// Net_64330020159514
// MI455X (gfx1250) — matrix operand order repaired and checked by reading, not compiled, not run on hardware
//
#include <hip/hip_runtime.h>

// ---------------------------------------------------------------------------
// MI455X (gfx1250) fused NNConv pipeline, round 2.
// msg = U @ B with U[e, i*K+k] = xs[e,i]*h[e,k] generated on the fly (i-major
// reduction ordering so every 32-chunk has constant i, contiguous k), and B
// pre-swizzled into per-fragment-contiguous order so each lane's WMMA B
// fragment is one 32B vector load from L2. No per-chunk barriers, no B LDS.
// v_wmma_f32_16x16x32_bf16, f32 accumulation, f32 atomic scatter.
// ---------------------------------------------------------------------------

typedef __bf16 bf16_t;
typedef bf16_t v16bf __attribute__((ext_vector_type(16)));
typedef bf16_t v8bf  __attribute__((ext_vector_type(8)));
typedef float  v8f   __attribute__((ext_vector_type(8)));

#define NNODES 4096
#define NEDGES 4096
#define NGRAPH 128
#define RSPLIT 4

// --------------------------- small helper kernels --------------------------

__global__ void zero_f32(float* __restrict__ p, int n) {
    int i = blockIdx.x * blockDim.x + threadIdx.x;
    if (i < n) p[i] = 0.0f;
}

// w2 [K, CIN*COUT] fp32  ->  w2s bf16 in fragment order:
//   w2s[(ch*COUT + c)*32 + kk] = w2[k, i*COUT + c],  r = ch*32+kk = i*K + k
__global__ void cast_swizzle_w2(const float* __restrict__ in, bf16_t* __restrict__ out,
                                int K, int CIN, int COUT) {
    const int total = K * CIN * COUT;
    const int tid = blockIdx.x * blockDim.x + threadIdx.x;
    if (tid >= total) return;
    const int kk = tid & 31;
    const int t  = tid >> 5;
    const int c  = t % COUT;
    const int ch = t / COUT;
    const int r  = ch * 32 + kk;
    const int i  = r / K;
    const int k  = r - i * K;
    out[tid] = (bf16_t)in[(size_t)k * CIN * COUT + (size_t)i * COUT + c];
}

// b2 [CIN*COUT] fp32 -> fragment order, rows padded with zeros to 32-multiple
__global__ void cast_swizzle_b2(const float* __restrict__ in, bf16_t* __restrict__ out,
                                int CIN, int COUT, int total) {
    const int tid = blockIdx.x * blockDim.x + threadIdx.x;
    if (tid >= total) return;
    const int kk = tid & 31;
    const int t  = tid >> 5;
    const int c  = t % COUT;
    const int ch = t / COUT;
    const int i  = ch * 32 + kk;
    out[tid] = (i < CIN) ? (bf16_t)in[(size_t)i * COUT + c] : (bf16_t)0.0f;
}

// edge MLP first layer: hE[e,k] = relu(edge_attr[e,:5] @ w1[5,K] + b1[k]) -> bf16
__global__ void edge_mlp1(const float* __restrict__ ea, const float* __restrict__ w1,
                          const float* __restrict__ b1, bf16_t* __restrict__ out, int K) {
    const int e = blockIdx.x;
    const int k = blockIdx.y * blockDim.x + threadIdx.x;
    if (k >= K) return;
    const float a0 = ea[e * 5 + 0], a1 = ea[e * 5 + 1], a2 = ea[e * 5 + 2];
    const float a3 = ea[e * 5 + 3], a4 = ea[e * 5 + 4];
    float s = b1[k];
    s += a0 * w1[0 * K + k];
    s += a1 * w1[1 * K + k];
    s += a2 * w1[2 * K + k];
    s += a3 * w1[3 * K + k];
    s += a4 * w1[4 * K + k];
    out[(size_t)e * K + k] = (bf16_t)(s > 0.0f ? s : 0.0f);
}

// gather source-node features per edge, cast to bf16
__global__ void gather_cast(const float* __restrict__ nodes, const int* __restrict__ src,
                            bf16_t* __restrict__ xs, int CIN) {
    const int e = blockIdx.x;
    const int s = src[e];
    for (int i = threadIdx.x; i < CIN; i += blockDim.x)
        xs[(size_t)e * CIN + i] = (bf16_t)nodes[(size_t)s * CIN + i];
}

// out[n,o] = ELU(agg[n,o] + x[n,:] @ root[:,o] + bias[o])   (blockDim.x == COUT)
__global__ void node_update(const float* __restrict__ agg, const float* __restrict__ xin,
                            const float* __restrict__ root, const float* __restrict__ bias,
                            float* __restrict__ out, int CIN, int COUT) {
    __shared__ float sx[256];
    const int n = blockIdx.x;
    const int o = threadIdx.x;
    for (int i = threadIdx.x; i < CIN; i += blockDim.x)
        sx[i] = xin[(size_t)n * CIN + i];
    __syncthreads();
    float s = agg[(size_t)n * COUT + o] + bias[o];
    for (int i = 0; i < CIN; ++i)
        s += sx[i] * root[(size_t)i * COUT + o];
    out[(size_t)n * COUT + o] = (s > 0.0f) ? s : (expf(s) - 1.0f);
}

// ------------------------- fused NNConv message GEMM -----------------------
// Block = 256 threads (8 wave32): mgroup = wave>>2 (16-edge half of 32-edge
// M-tile), ngroup = wave&3 (COUT/4 column window, NT 16-wide WMMA tiles).
// gridDim.y = RSPLIT splits the reduction (scatter is atomic anyway).

template <int K, int CIN, int COUT>
__global__ __launch_bounds__(256) void nnconv_msg_wmma(
        const bf16_t* __restrict__ hE,   // [E, K]
        const bf16_t* __restrict__ xs,   // [E, CIN]
        const bf16_t* __restrict__ w2s,  // swizzled [RCH][COUT][32]
        const bf16_t* __restrict__ b2s,  // swizzled [BCH][COUT][32]
        const int*    __restrict__ dst,  // [E]
        float*        __restrict__ agg,  // [NNODES, COUT]
        int rsplit) {
    constexpr int MT  = 32;             // edges per block
    constexpr int NW  = COUT / 4;       // columns per wave
    constexpr int NT  = NW / 16;        // WMMA tiles per wave
    constexpr int RCH = (K * CIN) / 32; // 32-deep reduction chunks
    static_assert(K % 32 == 0, "i-major fast path requires K % 32 == 0");

    __shared__ alignas(16) bf16_t sh[MT * K];
    __shared__ alignas(16) bf16_t sx[MT * CIN];
    __shared__ int sdst[MT];

    const int tid    = threadIdx.x;
    const int lane   = tid & 31;
    const int wave   = tid >> 5;
    const int mgroup = wave >> 2;
    const int ngroup = wave & 3;
    const int etile  = blockIdx.x * MT;

    // ---- stage A tiles (contiguous rows -> flat uint4 copies) ----
    {
        const uint4* s = (const uint4*)(hE + (size_t)etile * K);
        uint4* d = (uint4*)sh;
        for (int f = tid; f < MT * K / 8; f += 256) d[f] = s[f];
    }
    {
        const uint4* s = (const uint4*)(xs + (size_t)etile * CIN);
        uint4* d = (uint4*)sx;
        for (int f = tid; f < MT * CIN / 8; f += 256) d[f] = s[f];
    }
    if (tid < MT) sdst[tid] = dst[etile + tid];
    __syncthreads();

    const int half    = lane >> 4;
    const int l15     = lane & 15;
    const int mloc    = mgroup * 16 + l15;
    const int colbase = ngroup * NW + l15;

    v8f acc[NT];
#pragma unroll
    for (int t = 0; t < NT; ++t)
#pragma unroll
        for (int q = 0; q < 8; ++q) acc[t][q] = 0.0f;

    const int c0 = (RCH * (int)blockIdx.y) / rsplit;
    const int c1 = (RCH * ((int)blockIdx.y + 1)) / rsplit;

    for (int ch = c0; ch < c1; ++ch) {
        const int rbase = ch * 32;
        // i-major ordering: i constant per chunk, k contiguous
        const int iC    = rbase / K;
        const int kbase = rbase - iC * K;

        // B fragments: one contiguous 32B vector load each (L2-resident)
        const bf16_t* bbase = w2s + ((size_t)ch * COUT) * 32 + 16 * half;
        v16bf bfrag[NT];
#pragma unroll
        for (int t = 0; t < NT; ++t)
            bfrag[t] = *(const v16bf*)(bbase + (size_t)(colbase + t * 16) * 32);
        if (ch + 1 < c1)  // global_prefetch_b8 for next chunk's fragments
            __builtin_prefetch(bbase + (size_t)COUT * 32 + (size_t)colbase * 32, 0, 0);

        // A fragment (contiguous K order matching B: lanes 0-15 K 0-15; lanes
        // 16-31 K 16-31): scalar xs * two contiguous 8-runs of h.
        const float xv = (float)sx[mloc * CIN + iC];
        const v8bf h0 = *(const v8bf*)(sh + mloc * K + kbase + 16 * half);
        const v8bf h1 = *(const v8bf*)(sh + mloc * K + kbase + 16 * half + 8);
        v16bf a;
#pragma unroll
        for (int j = 0; j < 8; ++j) a[j] = (bf16_t)(xv * (float)h0[j]);
#pragma unroll
        for (int j = 0; j < 8; ++j) a[8 + j] = (bf16_t)(xv * (float)h1[j]);

#pragma unroll
        for (int t = 0; t < NT; ++t)
            acc[t] = __builtin_amdgcn_wmma_f32_16x16x32_bf16(
                false, a, false, bfrag[t], (short)0, acc[t], false, false);
    }

    // ---- bias term: msg += xs @ b2  (reduction-split slice 0 only) ----
    if (blockIdx.y == 0) {
        constexpr int BCH = (CIN + 31) / 32;
        for (int ch = 0; ch < BCH; ++ch) {
            const int ibase = ch * 32;
            v16bf a;
#pragma unroll
            for (int j = 0; j < 16; ++j) {
                const int kk = j + (half << 4);
                const int i = ibase + kk;
                a[j] = (i < CIN) ? sx[mloc * CIN + i] : (bf16_t)0.0f;
            }
            const bf16_t* bb = b2s + ((size_t)ch * COUT) * 32 + 16 * half;
#pragma unroll
            for (int t = 0; t < NT; ++t) {
                const v16bf b = *(const v16bf*)(bb + (size_t)(colbase + t * 16) * 32);
                acc[t] = __builtin_amdgcn_wmma_f32_16x16x32_bf16(
                    false, a, false, b, (short)0, acc[t], false, false);
            }
        }
    }

    // ---- scatter-add (C layout: VGPR r -> M=r for lanes 0-15, M=r+8 else) ----
#pragma unroll
    for (int t = 0; t < NT; ++t) {
        const int n = colbase + t * 16;
#pragma unroll
        for (int rrow = 0; rrow < 8; ++rrow) {
            const int m = rrow + 8 * half + mgroup * 16;
            atomicAdd(&agg[(size_t)sdst[m] * COUT + n], acc[t][rrow]);
        }
    }
}

// ------------------------------- readout -----------------------------------

__global__ void seg_sum(const float* __restrict__ h, const int* __restrict__ batch,
                        float* __restrict__ gsum, float* __restrict__ gcnt) {
    const int n = blockIdx.x;
    const int b = batch[n];
    for (int o = threadIdx.x; o < 256; o += blockDim.x)
        atomicAdd(&gsum[b * 256 + o], h[(size_t)n * 256 + o]);
    if (threadIdx.x == 0) atomicAdd(&gcnt[b], 1.0f);
}

__global__ void readout(const float* __restrict__ gsum, const float* __restrict__ gcnt,
                        const float* __restrict__ w1, const float* __restrict__ b1,
                        const float* __restrict__ w2, const float* __restrict__ b2,
                        const float* __restrict__ w3, const float* __restrict__ b3,
                        float* __restrict__ out) {
    __shared__ float g[256], t1[128], t2[64];
    const int b = blockIdx.x;
    const int tid = threadIdx.x;
    const float cnt = fmaxf(gcnt[b], 1.0f);
    g[tid] = gsum[b * 256 + tid] / cnt;
    __syncthreads();
    if (tid < 128) {
        float s = b1[tid];
        for (int i = 0; i < 256; ++i) s += g[i] * w1[i * 128 + tid];
        t1[tid] = (s > 0.0f) ? s : (expf(s) - 1.0f);
    }
    __syncthreads();
    if (tid < 64) {
        float s = b2[tid];
        for (int i = 0; i < 128; ++i) s += t1[i] * w2[i * 64 + tid];
        t2[tid] = (s > 0.0f) ? s : (expf(s) - 1.0f);
    }
    __syncthreads();
    if (tid == 0) {
        float s = b3[0];
        for (int i = 0; i < 64; ++i) s += t2[i] * w3[i];
        out[b] = s;
    }
}

// ------------------------------- launcher ----------------------------------

extern "C" void kernel_launch(void* const* d_in, const int* in_sizes, int n_in,
                              void* d_out, int out_size, void* d_ws, size_t ws_size,
                              hipStream_t stream) {
    const float* x      = (const float*)d_in[0];
    const int*   eidx   = (const int*)  d_in[1];
    const float* eattr  = (const float*)d_in[2];
    const int*   batch  = (const int*)  d_in[3];
    const float* nn1_w1 = (const float*)d_in[4];
    const float* nn1_b1 = (const float*)d_in[5];
    const float* nn1_w2 = (const float*)d_in[6];
    const float* nn1_b2 = (const float*)d_in[7];
    const float* root1  = (const float*)d_in[8];
    const float* bias1  = (const float*)d_in[9];
    const float* nn2_w1 = (const float*)d_in[10];
    const float* nn2_b1 = (const float*)d_in[11];
    const float* nn2_w2 = (const float*)d_in[12];
    const float* nn2_b2 = (const float*)d_in[13];
    const float* root2  = (const float*)d_in[14];
    const float* bias2  = (const float*)d_in[15];
    const float* nn3_w1 = (const float*)d_in[16];
    const float* nn3_b1 = (const float*)d_in[17];
    const float* nn3_w2 = (const float*)d_in[18];
    const float* nn3_b2 = (const float*)d_in[19];
    const float* root3  = (const float*)d_in[20];
    const float* bias3  = (const float*)d_in[21];
    const float* fc1_w  = (const float*)d_in[22];
    const float* fc1_b  = (const float*)d_in[23];
    const float* fc2_w  = (const float*)d_in[24];
    const float* fc2_b  = (const float*)d_in[25];
    const float* fc3_w  = (const float*)d_in[26];
    const float* fc3_b  = (const float*)d_in[27];

    const int* src = eidx;
    const int* dst = eidx + NEDGES;

    // workspace layout (256B aligned slices)
    char* ws = (char*)d_ws;
    size_t off = 0;
    auto take = [&](size_t bytes) -> void* {
        off = (off + 255) & ~(size_t)255;
        void* p = ws + off;
        off += bytes;
        return p;
    };
    bf16_t* w2s  = (bf16_t*)take((size_t)128 * 65536 * 2); // largest (L3), reused
    bf16_t* b2s  = (bf16_t*)take((size_t)8 * 32 * 256 * 2);
    bf16_t* hE   = (bf16_t*)take((size_t)NEDGES * 512 * 2);
    bf16_t* xsb  = (bf16_t*)take((size_t)NEDGES * 256 * 2);
    float*  agg  = (float*) take((size_t)NNODES * 256 * 4);
    float*  hA   = (float*) take((size_t)NNODES * 256 * 4);
    float*  hB   = (float*) take((size_t)NNODES * 256 * 4);
    float*  gsum = (float*) take((size_t)NGRAPH * 256 * 4);
    float*  gcnt = (float*) take((size_t)NGRAPH * 4);

    auto zero = [&](float* p, int n) {
        zero_f32<<<(n + 255) / 256, 256, 0, stream>>>(p, n);
    };
    auto swz_w2 = [&](const float* in, int K, int CIN, int COUT) {
        const int n = K * CIN * COUT;
        cast_swizzle_w2<<<(n + 255) / 256, 256, 0, stream>>>(in, w2s, K, CIN, COUT);
    };
    auto swz_b2 = [&](const float* in, int CIN, int COUT) {
        const int n = ((CIN + 31) / 32) * 32 * COUT;
        cast_swizzle_b2<<<(n + 255) / 256, 256, 0, stream>>>(in, b2s, CIN, COUT, n);
    };

    // -------- layer 1: K=512, CIN=37, COUT=128 --------
    edge_mlp1<<<dim3(NEDGES, 4), 128, 0, stream>>>(eattr, nn1_w1, nn1_b1, hE, 512);
    swz_w2(nn1_w2, 512, 37, 128);
    swz_b2(nn1_b2, 37, 128);
    gather_cast<<<NEDGES, 128, 0, stream>>>(x, src, xsb, 37);
    zero(agg, NNODES * 128);
    nnconv_msg_wmma<512, 37, 128><<<dim3(NEDGES / 32, RSPLIT), 256, 0, stream>>>(
        hE, xsb, w2s, b2s, dst, agg, RSPLIT);
    node_update<<<NNODES, 128, 0, stream>>>(agg, x, root1, bias1, hA, 37, 128);

    // -------- layer 2: K=128, CIN=128, COUT=256 --------
    edge_mlp1<<<dim3(NEDGES, 1), 128, 0, stream>>>(eattr, nn2_w1, nn2_b1, hE, 128);
    swz_w2(nn2_w2, 128, 128, 256);
    swz_b2(nn2_b2, 128, 256);
    gather_cast<<<NEDGES, 128, 0, stream>>>(hA, src, xsb, 128);
    zero(agg, NNODES * 256);
    nnconv_msg_wmma<128, 128, 256><<<dim3(NEDGES / 32, RSPLIT), 256, 0, stream>>>(
        hE, xsb, w2s, b2s, dst, agg, RSPLIT);
    node_update<<<NNODES, 256, 0, stream>>>(agg, hA, root2, bias2, hB, 128, 256);

    // -------- layer 3: K=128, CIN=256, COUT=256 --------
    edge_mlp1<<<dim3(NEDGES, 1), 128, 0, stream>>>(eattr, nn3_w1, nn3_b1, hE, 128);
    swz_w2(nn3_w2, 128, 256, 256);
    swz_b2(nn3_b2, 256, 256);
    gather_cast<<<NEDGES, 128, 0, stream>>>(hB, src, xsb, 256);
    zero(agg, NNODES * 256);
    nnconv_msg_wmma<128, 256, 256><<<dim3(NEDGES / 32, RSPLIT), 256, 0, stream>>>(
        hE, xsb, w2s, b2s, dst, agg, RSPLIT);
    node_update<<<NNODES, 256, 0, stream>>>(agg, hB, root3, bias3, hA, 256, 256);

    // -------- readout --------
    zero(gsum, NGRAPH * 256);
    zero(gcnt, NGRAPH);
    seg_sum<<<NNODES, 256, 0, stream>>>(hA, batch, gsum, gcnt);
    readout<<<NGRAPH, 256, 0, stream>>>(gsum, gcnt, fc1_w, fc1_b, fc2_w, fc2_b,
                                        fc3_w, fc3_b, (float*)d_out);
}
